// BipGraphNetwork_59785944760655
// MI455X (gfx1250) — compile-verified, matrix operand order checked by reading, not run on hardware
//
#include <hip/hip_runtime.h>
#include <hip/hip_bf16.h>
#include <math.h>

// Problem constants (match reference)
constexpr int kL  = 2;
constexpr int kNS = 768;
constexpr int kNT = 512;
constexpr int kF  = 128;
#define LEAK   0.01f
#define EPS_BN 1e-5f
#define EPS_N  1e-12f

typedef __attribute__((ext_vector_type(16))) _Float16 v16h;
typedef __attribute__((ext_vector_type(8)))  _Float16 h8;
typedef __attribute__((ext_vector_type(8)))  float    v8f;

#define WMMA_F16(a, b, c) \
    __builtin_amdgcn_wmma_f32_16x16x32_f16(false, (a), false, (b), (short)0, (c), false, false)

__device__ __forceinline__ v16h cat8(h8 lo, h8 hi) {
    return __builtin_shufflevector(lo, hi, 0, 1, 2, 3, 4, 5, 6, 7, 8, 9, 10, 11, 12, 13, 14, 15);
}
__device__ __forceinline__ float lrelu(float x) { return x >= 0.f ? x : LEAK * x; }

// ---------------------------------------------------------------------------
// Prep: copy layer-0 features, zero colsum accumulators, convert weights to
// f16, fold BatchNorm into per-channel scale/shift.  Runs every call so the
// launch is graph-replay safe (colsum atomics restart from zero).
// ---------------------------------------------------------------------------
__global__ __launch_bounds__(256) void prep_kernel(
    const float* __restrict__ fs_in, const float* __restrict__ ft_in,
    const float* __restrict__ e0w,
    const float* __restrict__ e0g, const float* __restrict__ e0b,
    const float* __restrict__ e0m, const float* __restrict__ e0v,
    const float* __restrict__ e1w,
    const float* __restrict__ e1g, const float* __restrict__ e1b,
    const float* __restrict__ e1m, const float* __restrict__ e1v,
    const float* __restrict__ n0w,
    const float* __restrict__ n0g, const float* __restrict__ n0b,
    const float* __restrict__ n0m, const float* __restrict__ n0v,
    const float* __restrict__ n1w,
    const float* __restrict__ n1g, const float* __restrict__ n1b,
    const float* __restrict__ n1m, const float* __restrict__ n1v,
    float* __restrict__ fs0, float* __restrict__ ft0, float* __restrict__ colsum,
    _Float16* __restrict__ e0w16, _Float16* __restrict__ e1w16,
    _Float16* __restrict__ n0w16, _Float16* __restrict__ n1w16,
    float* __restrict__ eb0sc, float* __restrict__ eb0sh,
    float* __restrict__ eb1sc, float* __restrict__ eb1sh,
    float* __restrict__ nb0sc, float* __restrict__ nb0sh,
    float* __restrict__ nb1sc, float* __restrict__ nb1sh)
{
    int tid = blockIdx.x * blockDim.x + threadIdx.x;
    int stride = gridDim.x * blockDim.x;
    for (int i = tid; i < kNS * kF; i += stride) fs0[i] = fs_in[i];
    for (int i = tid; i < kNT * kF; i += stride) ft0[i] = ft_in[i];
    for (int i = tid; i < kL * kNT; i += stride) colsum[i] = 0.f;
    for (int i = tid; i < kL * 256 * 128; i += stride) e0w16[i] = (_Float16)e0w[i];
    for (int i = tid; i < kL * 128 * 256; i += stride) e1w16[i] = (_Float16)e1w[i];
    for (int i = tid; i < kL * 256 * 256; i += stride) n0w16[i] = (_Float16)n0w[i];
    for (int i = tid; i < kL * 128 * 256; i += stride) n1w16[i] = (_Float16)n1w[i];
    for (int i = tid; i < kL * 256; i += stride) {
        float sc = e0g[i] * rsqrtf(e0v[i] + EPS_BN);
        eb0sc[i] = sc; eb0sh[i] = e0b[i] - e0m[i] * sc;
    }
    for (int i = tid; i < kL * 128; i += stride) {
        float sc = e1g[i] * rsqrtf(e1v[i] + EPS_BN);
        eb1sc[i] = sc; eb1sh[i] = e1b[i] - e1m[i] * sc;
    }
    for (int i = tid; i < kL * 256; i += stride) {
        float sc = n0g[i] * rsqrtf(n0v[i] + EPS_BN);
        nb0sc[i] = sc; nb0sh[i] = n0b[i] - n0m[i] * sc;
    }
    for (int i = tid; i < kL * 128; i += stride) {
        float sc = n1g[i] * rsqrtf(n1v[i] + EPS_BN);
        nb1sc[i] = sc; nb1sh[i] = n1b[i] - n1m[i] * sc;
    }
}

// ---------------------------------------------------------------------------
// Edge kernel: one wave = one (s, 16-t) pair tile, fused |diff| -> conv0(256)
// -> BN/lrelu -> conv1(128) -> BN/lrelu -> dot(w_out) -> sigmoid.
// Writes sim[s][t] and atomically accumulates per-t column sums.
// 4 waves / block; hbuf is per-wave LDS for the conv0->conv1 relayout.
// ---------------------------------------------------------------------------
__global__ __launch_bounds__(128) void edge_kernel(
    const float* __restrict__ fs, const float* __restrict__ ft,
    const _Float16* __restrict__ w0, const float* __restrict__ b0sc, const float* __restrict__ b0sh,
    const _Float16* __restrict__ w1, const float* __restrict__ b1sc, const float* __restrict__ b1sh,
    const float* __restrict__ wout, const float* __restrict__ outb,
    float* __restrict__ sim, float* __restrict__ colsum)
{
    __shared__ alignas(16) _Float16 hbuf[4][16][264];  // 16 pairs x 256ch, padded
    const int w    = threadIdx.x >> 5;
    const int lane = threadIdx.x & 31;
    const int ln   = lane & 15;
    const int hi   = lane >> 4;
    const int tile = blockIdx.x * 4 + w;
    const int s    = tile >> 5;           // 768 source rows
    const int t0   = (tile & 31) << 4;    // 32 t-tiles of 16

    const float* nsrow = fs + (size_t)s * kF;
    const float* ntrow = ft + (size_t)(t0 + ln) * kF;

    // A fragments for conv0: x[m=t, k] = |ns[k] - nt[t][k]|, K = 128 (4 chunks)
    v16h a[4];
#pragma unroll
    for (int c = 0; c < 4; c++) {
        int base = c * 32 + hi * 8;
        v8f sa = *(const v8f*)(nsrow + base);
        v8f sb = *(const v8f*)(nsrow + base + 16);
        v8f ta = *(const v8f*)(ntrow + base);
        v8f tb = *(const v8f*)(ntrow + base + 16);
        h8 lo, hv;
#pragma unroll
        for (int i = 0; i < 8; i++) {
            lo[i] = (_Float16)fabsf(sa[i] - ta[i]);
            hv[i] = (_Float16)fabsf(sb[i] - tb[i]);
        }
        a[c] = cat8(lo, hv);
    }

    // conv0: 256 output channels = 16 N-tiles, K = 128
#pragma unroll
    for (int j = 0; j < 16; j++) {
        const _Float16* wr = w0 + (size_t)(j * 16 + ln) * 128;
        v8f acc = {};
#pragma unroll
        for (int c = 0; c < 4; c++) {
            int base = c * 32 + hi * 8;
            v16h b = cat8(*(const h8*)(wr + base), *(const h8*)(wr + base + 16));
            acc = WMMA_F16(a[c], b, acc);
        }
        int ch = j * 16 + ln;
        float sc = b0sc[ch], sh = b0sh[ch];
#pragma unroll
        for (int r = 0; r < 8; r++) {
            float v = lrelu(acc[r] * sc + sh);
            hbuf[w][r + 8 * hi][ch] = (_Float16)v;
        }
    }
    __syncthreads();

    // conv1 A fragments from LDS: K = 256 (8 chunks)
    v16h a2[8];
#pragma unroll
    for (int c = 0; c < 8; c++) {
        int base = c * 32 + hi * 8;
        a2[c] = cat8(*(const h8*)(&hbuf[w][ln][base]),
                     *(const h8*)(&hbuf[w][ln][base + 16]));
    }

    float part[8];
#pragma unroll
    for (int r = 0; r < 8; r++) part[r] = 0.f;

    // conv1: 128 output channels = 8 N-tiles; fused dot with w_out
#pragma unroll
    for (int j = 0; j < 8; j++) {
        const _Float16* wr = w1 + (size_t)(j * 16 + ln) * 256;
        v8f acc = {};
#pragma unroll
        for (int c = 0; c < 8; c++) {
            int base = c * 32 + hi * 8;
            v16h b = cat8(*(const h8*)(wr + base), *(const h8*)(wr + base + 16));
            acc = WMMA_F16(a2[c], b, acc);
        }
        int ch = j * 16 + ln;
        float sc = b1sc[ch], sh = b1sh[ch], wo = wout[ch];
#pragma unroll
        for (int r = 0; r < 8; r++) part[r] += lrelu(acc[r] * sc + sh) * wo;
    }

    // reduce the 128-wide dot across the 16 lanes sharing each pair row
#pragma unroll
    for (int m = 1; m < 16; m <<= 1) {
#pragma unroll
        for (int r = 0; r < 8; r++) part[r] += __shfl_xor(part[r], m, 32);
    }
    float bb = outb[0];
    if (ln == 0) {
#pragma unroll
        for (int r = 0; r < 8; r++) {
            int t = t0 + r + 8 * hi;
            float sg = 1.0f / (1.0f + __expf(-(part[r] + bb)));
            sim[(size_t)s * kNT + t] = sg;
            atomicAdd(&colsum[t], sg);
        }
    }
}

// ---------------------------------------------------------------------------
// Normalize: a = sim/colsum, edge = a/rowsum(a). One block per source row.
// Writes final edge output plus edge and edge^T workspace copies.
// ---------------------------------------------------------------------------
__global__ __launch_bounds__(256) void norm_kernel(
    const float* __restrict__ sim, const float* __restrict__ colsum,
    float* __restrict__ edge_out, float* __restrict__ edge_ws,
    float* __restrict__ edgeT_ws)
{
    __shared__ float red[256];
    int s = blockIdx.x, tid = threadIdx.x;
    float a0 = sim[(size_t)s * kNT + tid]       / fmaxf(colsum[tid],       EPS_N);
    float a1 = sim[(size_t)s * kNT + tid + 256] / fmaxf(colsum[tid + 256], EPS_N);
    red[tid] = a0 + a1;
    __syncthreads();
    for (int o = 128; o > 0; o >>= 1) {
        if (tid < o) red[tid] += red[tid + o];
        __syncthreads();
    }
    float inv = 1.0f / fmaxf(red[0], EPS_N);
    float e0 = a0 * inv, e1 = a1 * inv;
    edge_out[(size_t)s * kNT + tid]        = e0;
    edge_out[(size_t)s * kNT + tid + 256]  = e1;
    edge_ws [(size_t)s * kNT + tid]        = e0;
    edge_ws [(size_t)s * kNT + tid + 256]  = e1;
    edgeT_ws[(size_t)tid * kNS + s]         = e0;
    edgeT_ws[(size_t)(tid + 256) * kNS + s] = e1;
}

// ---------------------------------------------------------------------------
// Aggregation GEMM: out[M][128] = E[M][K] @ Fm[K][128] via f16 WMMA.
// Block = 8 waves x one 16-row M-tile each; Fm chunk staged transposed in LDS.
// ---------------------------------------------------------------------------
__global__ __launch_bounds__(256) void aggr_kernel(
    const float* __restrict__ E, const float* __restrict__ Fm,
    float* __restrict__ out, int K)
{
    __shared__ alignas(16) _Float16 ldsT[128][40];  // Fm chunk, transposed + padded
    const int w    = threadIdx.x >> 5;
    const int lane = threadIdx.x & 31;
    const int ln   = lane & 15;
    const int hi   = lane >> 4;
    const int m0   = (blockIdx.x * 8 + w) * 16;

    v8f acc[8] = {};
    const int nk = K >> 5;
    for (int kc = 0; kc < nk; kc++) {
        __syncthreads();
        for (int idx = threadIdx.x; idx < 32 * 128; idx += 256) {
            int r = idx >> 7, c = idx & 127;
            ldsT[c][r] = (_Float16)Fm[(size_t)(kc * 32 + r) * kF + c];
        }
        __syncthreads();

        const float* erow = E + (size_t)(m0 + ln) * K + kc * 32 + hi * 8;
        v8f lo = *(const v8f*)erow;
        v8f hv = *(const v8f*)(erow + 16);
        h8 alo, ahi;
#pragma unroll
        for (int i = 0; i < 8; i++) { alo[i] = (_Float16)lo[i]; ahi[i] = (_Float16)hv[i]; }
        v16h a = cat8(alo, ahi);
#pragma unroll
        for (int j = 0; j < 8; j++) {
            const _Float16* bp = &ldsT[16 * j + ln][hi * 8];
            v16h b = cat8(*(const h8*)bp, *(const h8*)(bp + 16));
            acc[j] = WMMA_F16(a, b, acc[j]);
        }
    }
#pragma unroll
    for (int j = 0; j < 8; j++) {
#pragma unroll
        for (int r = 0; r < 8; r++)
            out[(size_t)(m0 + r + 8 * hi) * kF + 16 * j + ln] = acc[j][r];
    }
}

// ---------------------------------------------------------------------------
// Node MLP: z = [feat, aggr] (256ch) -> conv0(256) -> BN/lrelu -> conv1(128)
// -> BN/lrelu. One wave per 16-row tile; 4 waves per block.
// Writes both workspace (next layer input) and final output slice.
// ---------------------------------------------------------------------------
__global__ __launch_bounds__(128) void node_kernel(
    const float* __restrict__ feat, const float* __restrict__ aggr,
    const _Float16* __restrict__ w0, const float* __restrict__ b0sc, const float* __restrict__ b0sh,
    const _Float16* __restrict__ w1, const float* __restrict__ b1sc, const float* __restrict__ b1sh,
    float* __restrict__ out_ws, float* __restrict__ out_final)
{
    __shared__ alignas(16) _Float16 hbuf[4][16][264];
    const int w    = threadIdx.x >> 5;
    const int lane = threadIdx.x & 31;
    const int ln   = lane & 15;
    const int hi   = lane >> 4;
    const int m0   = (blockIdx.x * 4 + w) * 16;
    const int row  = m0 + ln;

    // A fragments: chunks 0-3 from feat, 4-7 from aggr (concat along K)
    v16h a[8];
#pragma unroll
    for (int c = 0; c < 8; c++) {
        const float* src = (c < 4) ? (feat + (size_t)row * kF + c * 32)
                                   : (aggr + (size_t)row * kF + (c - 4) * 32);
        v8f lo = *(const v8f*)(src + hi * 8);
        v8f hv = *(const v8f*)(src + hi * 8 + 16);
        h8 alo, ahi;
#pragma unroll
        for (int i = 0; i < 8; i++) { alo[i] = (_Float16)lo[i]; ahi[i] = (_Float16)hv[i]; }
        a[c] = cat8(alo, ahi);
    }

    // conv0: 256 out channels, K = 256
#pragma unroll
    for (int j = 0; j < 16; j++) {
        const _Float16* wr = w0 + (size_t)(j * 16 + ln) * 256;
        v8f acc = {};
#pragma unroll
        for (int c = 0; c < 8; c++) {
            int base = c * 32 + hi * 8;
            v16h b = cat8(*(const h8*)(wr + base), *(const h8*)(wr + base + 16));
            acc = WMMA_F16(a[c], b, acc);
        }
        int ch = j * 16 + ln;
        float sc = b0sc[ch], sh = b0sh[ch];
#pragma unroll
        for (int r = 0; r < 8; r++) {
            float v = lrelu(acc[r] * sc + sh);
            hbuf[w][r + 8 * hi][ch] = (_Float16)v;
        }
    }
    __syncthreads();

    v16h a2[8];
#pragma unroll
    for (int c = 0; c < 8; c++) {
        int base = c * 32 + hi * 8;
        a2[c] = cat8(*(const h8*)(&hbuf[w][ln][base]),
                     *(const h8*)(&hbuf[w][ln][base + 16]));
    }

    // conv1: 128 out channels, K = 256
#pragma unroll
    for (int j = 0; j < 8; j++) {
        const _Float16* wr = w1 + (size_t)(j * 16 + ln) * 256;
        v8f acc = {};
#pragma unroll
        for (int c = 0; c < 8; c++) {
            int base = c * 32 + hi * 8;
            v16h b = cat8(*(const h8*)(wr + base), *(const h8*)(wr + base + 16));
            acc = WMMA_F16(a2[c], b, acc);
        }
        int ch = j * 16 + ln;
        float sc = b1sc[ch], sh = b1sh[ch];
#pragma unroll
        for (int r = 0; r < 8; r++) {
            float v = lrelu(acc[r] * sc + sh);
            size_t rr = (size_t)(m0 + r + 8 * hi) * kF + ch;
            out_ws[rr] = v;
            out_final[rr] = v;
        }
    }
}

// ---------------------------------------------------------------------------
extern "C" void kernel_launch(void* const* d_in, const int* in_sizes, int n_in,
                              void* d_out, int out_size, void* d_ws, size_t ws_size,
                              hipStream_t stream)
{
    const float* fs_in = (const float*)d_in[0];
    const float* ft_in = (const float*)d_in[1];
    const float* e0w = (const float*)d_in[2];
    const float* e0g = (const float*)d_in[3], *e0b = (const float*)d_in[4];
    const float* e0m = (const float*)d_in[5], *e0v = (const float*)d_in[6];
    const float* e1w = (const float*)d_in[7];
    const float* e1g = (const float*)d_in[8], *e1b = (const float*)d_in[9];
    const float* e1m = (const float*)d_in[10], *e1v = (const float*)d_in[11];
    const float* eow = (const float*)d_in[12];
    const float* eob = (const float*)d_in[13];
    const float* n0w = (const float*)d_in[14];
    const float* n0g = (const float*)d_in[15], *n0b = (const float*)d_in[16];
    const float* n0m = (const float*)d_in[17], *n0v = (const float*)d_in[18];
    const float* n1w = (const float*)d_in[19];
    const float* n1g = (const float*)d_in[20], *n1b = (const float*)d_in[21];
    const float* n1m = (const float*)d_in[22], *n1v = (const float*)d_in[23];

    char* ws = (char*)d_ws;
    size_t off = 0;
    auto alloc = [&](size_t bytes) -> char* {
        char* p = ws + off;
        off = (off + bytes + 255) & ~(size_t)255;
        return p;
    };
    float* fs     = (float*)alloc((size_t)3 * kNS * kF * 4);  // features per layer 0..2
    float* ftb    = (float*)alloc((size_t)3 * kNT * kF * 4);
    float* sim    = (float*)alloc((size_t)kNS * kNT * 4);
    float* colsum = (float*)alloc((size_t)kL * kNT * 4);
    float* edge   = (float*)alloc((size_t)kNS * kNT * 4);
    float* edgeT  = (float*)alloc((size_t)kNT * kNS * 4);
    float* aggr_s = (float*)alloc((size_t)kNS * kF * 4);
    float* aggr_t = (float*)alloc((size_t)kNT * kF * 4);
    _Float16* e0w16 = (_Float16*)alloc((size_t)kL * 256 * 128 * 2);
    _Float16* e1w16 = (_Float16*)alloc((size_t)kL * 128 * 256 * 2);
    _Float16* n0w16 = (_Float16*)alloc((size_t)kL * 256 * 256 * 2);
    _Float16* n1w16 = (_Float16*)alloc((size_t)kL * 128 * 256 * 2);
    float* eb0sc = (float*)alloc((size_t)kL * 256 * 4);
    float* eb0sh = (float*)alloc((size_t)kL * 256 * 4);
    float* eb1sc = (float*)alloc((size_t)kL * 128 * 4);
    float* eb1sh = (float*)alloc((size_t)kL * 128 * 4);
    float* nb0sc = (float*)alloc((size_t)kL * 256 * 4);
    float* nb0sh = (float*)alloc((size_t)kL * 256 * 4);
    float* nb1sc = (float*)alloc((size_t)kL * 128 * 4);
    float* nb1sh = (float*)alloc((size_t)kL * 128 * 4);

    float* out       = (float*)d_out;
    float* out_edges = out;                                // [L][NS][NT]
    float* out_srcs  = out + (size_t)kL * kNS * kNT;       // [L][NS][F]
    float* out_tgts  = out_srcs + (size_t)kL * kNS * kF;   // [L][NT][F]

    prep_kernel<<<256, 256, 0, stream>>>(
        fs_in, ft_in,
        e0w, e0g, e0b, e0m, e0v,
        e1w, e1g, e1b, e1m, e1v,
        n0w, n0g, n0b, n0m, n0v,
        n1w, n1g, n1b, n1m, n1v,
        fs, ftb, colsum,
        e0w16, e1w16, n0w16, n1w16,
        eb0sc, eb0sh, eb1sc, eb1sh,
        nb0sc, nb0sh, nb1sc, nb1sh);

    for (int l = 0; l < kL; l++) {
        const float* cfs = fs  + (size_t)l * kNS * kF;
        const float* cft = ftb + (size_t)l * kNT * kF;

        // 768 s * 32 t-tiles = 24576 wave-tiles, 4 waves per block
        edge_kernel<<<(kNS * (kNT / 16)) / 4, 128, 0, stream>>>(
            cfs, cft,
            e0w16 + (size_t)l * 256 * 128, eb0sc + l * 256, eb0sh + l * 256,
            e1w16 + (size_t)l * 128 * 256, eb1sc + l * 128, eb1sh + l * 128,
            eow + (size_t)l * 128, eob + l,
            sim, colsum + (size_t)l * kNT);

        norm_kernel<<<kNS, 256, 0, stream>>>(
            sim, colsum + (size_t)l * kNT,
            out_edges + (size_t)l * kNS * kNT, edge, edgeT);

        aggr_kernel<<<kNS / 128, 256, 0, stream>>>(edge,  cft, aggr_s, kNT);
        aggr_kernel<<<kNT / 128, 256, 0, stream>>>(edgeT, cfs, aggr_t, kNS);

        node_kernel<<<kNS / 64, 128, 0, stream>>>(
            cfs, aggr_s,
            n0w16 + (size_t)l * 256 * 256, nb0sc + l * 256, nb0sh + l * 256,
            n1w16 + (size_t)l * 128 * 256, nb1sc + l * 128, nb1sh + l * 128,
            fs + (size_t)(l + 1) * kNS * kF, out_srcs + (size_t)l * kNS * kF);

        node_kernel<<<kNT / 64, 128, 0, stream>>>(
            cft, aggr_t,
            n0w16 + (size_t)l * 256 * 256, nb0sc + l * 256, nb0sh + l * 256,
            n1w16 + (size_t)l * 128 * 256, nb1sc + l * 128, nb1sh + l * 128,
            ftb + (size_t)(l + 1) * kNT * kF, out_tgts + (size_t)l * kNT * kF);
    }
}
